// MaceNet_29961691857584
// MI455X (gfx1250) — matrix operand order repaired and checked by reading, not compiled, not run on hardware
//
#include <hip/hip_runtime.h>

// ---------------- CDNA5 WMMA types ----------------
typedef __attribute__((ext_vector_type(16))) __bf16 v16bf;
typedef __attribute__((ext_vector_type(8)))  float  v8f;

union v16bf_u {
    v16bf        v;
    unsigned int u[8];
};

__device__ __forceinline__ unsigned short f32_to_bf16(float f) {
    unsigned int u = __float_as_uint(f);
    unsigned int r = u + 0x7FFFu + ((u >> 16) & 1u);   // round-to-nearest-even
    return (unsigned short)(r >> 16);
}

// ---------------- problem constants ----------------
#define NB    8     // batch (graphs)
#define NN    128   // nodes
#define FS    128
#define FV    64
#define NBES  8
#define HID   64
#define ROUT  320   // FS + 3*FV
#define INV_AVG (1.0f/128.0f)

// =====================================================================
// k_prep: Wr2 (2,64,320) f32 -> WT (2,320,64) bf16 (transposed so that
// consecutive K are adjacent -> paired dword loads for B fragments)
// =====================================================================
__global__ void k_prep(const float* __restrict__ Wr2, unsigned short* __restrict__ WT) {
    int idx = blockIdx.x * blockDim.x + threadIdx.x;
    if (idx >= 2 * ROUT * HID) return;
    int l = idx / (ROUT * HID);
    int rem = idx - l * ROUT * HID;
    int n = rem >> 6;        // 0..319
    int k = rem & 63;        // 0..63
    WT[idx] = f32_to_bf16(Wr2[l * HID * ROUT + k * ROUT + n]);
}

// =====================================================================
// k_init: per-graph center of mass, s broadcast, V = 0
// =====================================================================
__global__ void k_init(const float* __restrict__ x, const float* __restrict__ species,
                       float* __restrict__ S, float* __restrict__ Vv, float* __restrict__ com) {
    int b = blockIdx.x;
    int t = threadIdx.x;
    if (t < 3) {
        float acc = 0.f;
        for (int n = 0; n < NN; ++n) acc += x[(b * NN + n) * 3 + t];
        com[b * 4 + t] = acc * (1.0f / (float)NN);
    }
    for (int i = t; i < NN * FS; i += blockDim.x)       S[b * NN * FS + i]       = species[i & (FS - 1)];
    for (int i = t; i < NN * FV * 3; i += blockDim.x)   Vv[b * NN * FV * 3 + i]  = 0.f;
}

// =====================================================================
// k_sw: SW[b] = S[b] @ W_sv[l]   (128x128x64 per graph, negligible FLOP;
// factoring this out of the per-edge formula saves ~260 MFLOP/graph/layer)
// =====================================================================
__global__ void k_sw(const float* __restrict__ S, const float* __restrict__ Wsv,
                     float* __restrict__ SW) {
    int b = blockIdx.x;
    for (int idx = threadIdx.x; idx < NN * FV; idx += blockDim.x) {
        int n = idx >> 6, g = idx & 63;
        const float* srow = &S[(b * NN + n) * FS];
        float acc = 0.f;
        #pragma unroll 8
        for (int f = 0; f < FS; ++f) acc += srow[f] * Wsv[f * FV + g];
        SW[(b * NN + n) * FV + g] = acc;
    }
}

// =====================================================================
// k_layer: FUSED  geometry -> bessel -> radial MLP (bf16 WMMA) ->
//          message formation -> receiver-side segment sum.
// One workgroup per (graph, receiver node); 4 waves; 127 senders in
// 8 tiles of 16 edges. h tile (16x320) lives only in LDS.
// =====================================================================
__global__ void __launch_bounds__(128)
k_layer(const float* __restrict__ x,
        const float* __restrict__ S,          // (B,N,128)
        const float* __restrict__ Vv,         // (B,N,64,3)
        const float* __restrict__ SW,         // (B,N,64)  = s @ W_sv
        const float* __restrict__ Wr1,        // (8,64)    layer slice
        const unsigned short* __restrict__ WT,// (320,64)  bf16, layer slice
        float* __restrict__ AGG0,             // (B,N,192)
        float* __restrict__ AGG1)             // (B,N,128,3)
{
    __shared__ float        sWr1[NBES * HID];     // 2 KB
    __shared__ float        sRB[16 * NBES];       // 16 edges x 8 bessel
    __shared__ float        sVH[16 * 4];          // vhat (xyz) per edge
    __shared__ int          sSend[16];
    __shared__ unsigned int sHidU[16 * HID / 2];  // 16x64 bf16 hidden
    __shared__ float        sHT[16 * ROUT];       // 16x320 f32 h-tile (20 KB)

    const int t     = threadIdx.x;
    const int b     = blockIdx.x >> 7;
    const int rnode = blockIdx.x & 127;
    const int lane  = t & 31;
    const int wv    = t >> 5;

    for (int i = t; i < NBES * HID; i += 128) sWr1[i] = Wr1[i];

    const float rx = x[(b * NN + rnode) * 3 + 0];
    const float ry = x[(b * NN + rnode) * 3 + 1];
    const float rz = x[(b * NN + rnode) * 3 + 2];

    // ---- preload B fragments: wave wv owns N-tiles [wv*5, wv*5+5) ----
    // B 32x16 bf16, contiguous k used here (matches A): lane: N = lane&15; VGPR q: K = (lane<16?0:16)+2q(,+1)
    const int nt0 = wv * 5;
    const unsigned int* WTu = (const unsigned int*)WT;
    v16bf_u bfr[5][2];
    {
        const int nlo = lane & 15;
        const int kb2h = (lane < 16) ? 0 : 8;   // kb2/2
        #pragma unroll
        for (int j = 0; j < 5; ++j) {
            const int ncol = (nt0 + j) * 16 + nlo;
            #pragma unroll
            for (int c = 0; c < 2; ++c) {
                const int base = ncol * 32 + c * 16 + kb2h;
                #pragma unroll
                for (int q = 0; q < 8; ++q) bfr[j][c].u[q] = WTu[base + q];
            }
        }
    }

    float acc_a = 0.f, acc_b = 0.f;
    float ax0 = 0.f, ax1 = 0.f, ax2 = 0.f;

    for (int T = 0; T < 8; ++T) {
        __syncthreads();   // previous tile's LDS fully consumed
        // ---------- phase A: geometry + bessel basis (16 threads) ----------
        if (t < 16) {
            int slot = T * 16 + t;
            int snd; float mask;
            if (slot >= NN - 1) { snd = rnode; mask = 0.f; }          // pad edge
            else { snd = slot + (slot >= rnode ? 1 : 0); mask = 1.f; }
            sSend[t] = snd;
            float vx = rx - x[(b * NN + snd) * 3 + 0];
            float vy = ry - x[(b * NN + snd) * 3 + 1];
            float vz = rz - x[(b * NN + snd) * 3 + 2];
            float r  = sqrtf(vx * vx + vy * vy + vz * vz) + 1e-8f;
            float inv = 1.0f / r;
            sVH[t * 4 + 0] = vx * inv;
            sVH[t * 4 + 1] = vy * inv;
            sVH[t * 4 + 2] = vz * inv;
            float u = r * 0.2f;                       // r / R_MAX
            float env = 0.f;
            if (u < 1.f) {
                float u2 = u * u, u6 = u2 * u2 * u2;
                env = 1.f - 28.f * u6 + 48.f * u6 * u - 21.f * u6 * u2;
            }
            float pref = 0.632455532f * inv * env * mask;   // sqrt(2/5)/r*env
            #pragma unroll
            for (int n = 0; n < NBES; ++n)
                sRB[t * NBES + n] = pref * __sinf((float)(n + 1) * 0.6283185307f * r);
        }
        __syncthreads();
        // ---------- phase B: hidden = silu(rb @ Wr1) -> bf16 LDS ----------
        {
            unsigned short* sHid = (unsigned short*)sHidU;
            const int i  = t & 15;
            const int j0 = (t >> 4) * 8;
            #pragma unroll
            for (int jj = 0; jj < 8; ++jj) {
                const int j = j0 + jj;
                float acc = 0.f;
                #pragma unroll
                for (int n = 0; n < NBES; ++n) acc += sRB[i * NBES + n] * sWr1[n * HID + j];
                float h = acc / (1.f + __expf(-acc));        // silu
                sHid[i * HID + j] = f32_to_bf16(h);
            }
        }
        __syncthreads();
        // ---------- phase C: WMMA  h_tile = hidden(16x64) @ Wr2(64x320) ----------
        {
            // A 16x32 bf16, contiguous k used here (matches B): lane: M=lane&15; kb=(lane<16?0:16);
            // VGPR q: K=kb+2q  (+1 in high half)
            const int M  = lane & 15;
            const int kb = (lane < 16) ? 0 : 16;
            v16bf_u afr[2];
            #pragma unroll
            for (int c = 0; c < 2; ++c) {
                #pragma unroll
                for (int q = 0; q < 8; ++q) {
                    int kp = c * 32 + kb + 2 * q;
                    afr[c].u[q] = sHidU[M * 32 + (kp >> 1)];
                }
            }
            const int nlo = lane & 15;
            const int mb  = (lane < 16) ? 0 : 8;
            #pragma unroll
            for (int j = 0; j < 5; ++j) {
                v8f cacc = {};
                cacc = __builtin_amdgcn_wmma_f32_16x16x32_bf16(
                           false, afr[0].v, false, bfr[j][0].v, (short)0, cacc, false, false);
                cacc = __builtin_amdgcn_wmma_f32_16x16x32_bf16(
                           false, afr[1].v, false, bfr[j][1].v, (short)0, cacc, false, false);
                const int ncol = (nt0 + j) * 16 + nlo;
                #pragma unroll
                for (int p = 0; p < 8; ++p) sHT[(mb + p) * ROUT + ncol] = cacc[p];
            }
        }
        __syncthreads();
        // ---------- phase D: message formation + receiver reduction ----------
        for (int i = 0; i < 16; ++i) {
            const int snd = sSend[i];
            if (t == 0 && i + 1 < 16)
                __builtin_prefetch(&S[(b * NN + sSend[i + 1]) * FS], 0, 1);
            const float* hrow = &sHT[i * ROUT];
            // m0a: R0 * s_src
            acc_a += hrow[t] * S[(b * NN + snd) * FS + t];
            const float vh0 = sVH[i * 4 + 0], vh1 = sVH[i * 4 + 1], vh2 = sVH[i * 4 + 2];
            if (t < FV) {
                const float* Vp = &Vv[((b * NN + snd) * FV + t) * 3];
                float vdot = Vp[0] * vh0 + Vp[1] * vh1 + Vp[2] * vh2;
                acc_b += hrow[FS + t] * vdot;                       // m0b
                float cm = hrow[FS + FV + t] * SW[(b * NN + snd) * FV + t]; // m1a coef
                ax0 += cm * vh0; ax1 += cm * vh1; ax2 += cm * vh2;
            } else {
                const int fv = t - FV;
                const float* Vp = &Vv[((b * NN + snd) * FV + fv) * 3];
                float r3 = hrow[FS + 2 * FV + fv];                  // m1b
                ax0 += r3 * Vp[0]; ax1 += r3 * Vp[1]; ax2 += r3 * Vp[2];
            }
        }
    }

    // ---------- write segment sums / AVG_NEIGH ----------
    AGG0[(b * NN + rnode) * (FS + FV) + t] = acc_a * INV_AVG;
    if (t < FV) AGG0[(b * NN + rnode) * (FS + FV) + FS + t] = acc_b * INV_AVG;
    float* a1 = &AGG1[((b * NN + rnode) * (2 * FV) + t) * 3];
    a1[0] = ax0 * INV_AVG; a1[1] = ax1 * INV_AVG; a1[2] = ax2 * INV_AVG;
}

// =====================================================================
// k_update: s += silu(agg0 @ W_mix_s);  V += einsum(agg1, W_mix_v)
// =====================================================================
__global__ void k_update(const float* __restrict__ AGG0, const float* __restrict__ AGG1,
                         const float* __restrict__ Wms, const float* __restrict__ Wmv,
                         float* __restrict__ S, float* __restrict__ Vv) {
    int b = blockIdx.x;
    for (int idx = threadIdx.x; idx < NN * FS; idx += blockDim.x) {
        int n = idx >> 7, g = idx & 127;
        const float* arow = &AGG0[(b * NN + n) * (FS + FV)];
        float acc = 0.f;
        #pragma unroll 8
        for (int f = 0; f < FS + FV; ++f) acc += arow[f] * Wms[f * FS + g];
        S[(b * NN + n) * FS + g] += acc / (1.f + __expf(-acc));
    }
    for (int idx = threadIdx.x; idx < NN * FV * 3; idx += blockDim.x) {
        int k = idx % 3; int rest = idx / 3;
        int g = rest & 63, n = rest >> 6;
        const float* arow = &AGG1[(b * NN + n) * (2 * FV) * 3];
        float acc = 0.f;
        #pragma unroll 8
        for (int f = 0; f < 2 * FV; ++f) acc += arow[f * 3 + k] * Wmv[f * FV + g];
        Vv[((b * NN + n) * FV + g) * 3 + k] += acc;
    }
}

// =====================================================================
// k_out: out_v = V @ W_out_v + com (first in d_out), out_s = S @ W_out_s
// =====================================================================
__global__ void k_out(const float* __restrict__ S, const float* __restrict__ Vv,
                      const float* __restrict__ Wos, const float* __restrict__ Wov,
                      const float* __restrict__ com, float* __restrict__ out) {
    int b = blockIdx.x;
    float* out_v = out;                               // (B,128,64,3)
    float* out_s = out + NB * NN * FV * 3;            // (B,128,128)
    for (int idx = threadIdx.x; idx < NN * FV * 3; idx += blockDim.x) {
        int k = idx % 3; int rest = idx / 3;
        int g = rest & 63, n = rest >> 6;
        float acc = com[b * 4 + k];
        #pragma unroll 8
        for (int f = 0; f < FV; ++f)
            acc += Vv[((b * NN + n) * FV + f) * 3 + k] * Wov[f * FV + g];
        out_v[((b * NN + n) * FV + g) * 3 + k] = acc;
    }
    for (int idx = threadIdx.x; idx < NN * FS; idx += blockDim.x) {
        int n = idx >> 7, g = idx & 127;
        const float* srow = &S[(b * NN + n) * FS];
        float acc = 0.f;
        #pragma unroll 8
        for (int f = 0; f < FS; ++f) acc += srow[f] * Wos[f * FS + g];
        out_s[(b * NN + n) * FS + g] = acc;
    }
}

// =====================================================================
extern "C" void kernel_launch(void* const* d_in, const int* in_sizes, int n_in,
                              void* d_out, int out_size, void* d_ws, size_t ws_size,
                              hipStream_t stream) {
    (void)in_sizes; (void)n_in; (void)out_size; (void)ws_size;
    const float* x    = (const float*)d_in[0];
    const float* spe  = (const float*)d_in[1];
    const float* Wr1  = (const float*)d_in[2];  // (2,8,64)
    const float* Wr2  = (const float*)d_in[3];  // (2,64,320)
    const float* Wsv  = (const float*)d_in[4];  // (2,128,64)
    const float* Wms  = (const float*)d_in[5];  // (2,192,128)
    const float* Wmv  = (const float*)d_in[6];  // (2,128,64)
    const float* Wos  = (const float*)d_in[7];  // (128,128)
    const float* Wov  = (const float*)d_in[8];  // (64,64)
    float* out = (float*)d_out;

    char* ws = (char*)d_ws;
    size_t off = 0;
    auto alloc = [&](size_t bytes) -> void* {
        void* p = ws + off;
        off += (bytes + 255) & ~(size_t)255;
        return p;
    };
    float* S    = (float*)alloc((size_t)NB * NN * FS * 4);
    float* Vv   = (float*)alloc((size_t)NB * NN * FV * 3 * 4);
    float* SW   = (float*)alloc((size_t)NB * NN * FV * 4);
    float* AGG0 = (float*)alloc((size_t)NB * NN * (FS + FV) * 4);
    float* AGG1 = (float*)alloc((size_t)NB * NN * (2 * FV) * 3 * 4);
    float* COM  = (float*)alloc((size_t)NB * 4 * 4);
    unsigned short* WT = (unsigned short*)alloc((size_t)2 * ROUT * HID * 2);

    k_prep<<<(2 * ROUT * HID + 255) / 256, 256, 0, stream>>>(Wr2, WT);
    k_init<<<NB, 256, 0, stream>>>(x, spe, S, Vv, COM);
    for (int l = 0; l < 2; ++l) {
        k_sw<<<NB, 256, 0, stream>>>(S, Wsv + (size_t)l * FS * FV, SW);
        k_layer<<<NB * NN, 128, 0, stream>>>(x, S, Vv, SW,
                                             Wr1 + (size_t)l * NBES * HID,
                                             WT + (size_t)l * ROUT * HID,
                                             AGG0, AGG1);
        k_update<<<NB, 256, 0, stream>>>(AGG0, AGG1,
                                         Wms + (size_t)l * (FS + FV) * FS,
                                         Wmv + (size_t)l * (2 * FV) * FV,
                                         S, Vv);
    }
    k_out<<<NB, 256, 0, stream>>>(S, Vv, Wos, Wov, COM, out);
}
